// psp_net_21517786153620
// MI455X (gfx1250) — matrix operand order repaired and checked by reading, not compiled, not run on hardware
//
#include <hip/hip_runtime.h>
#include <hip/hip_bf16.h>
#include <math.h>

// ---------------------------------------------------------------------------
// Types for CDNA5 WMMA (wave32): A,B are 16 x f16 per lane, C/D are 8 x f32.
// ---------------------------------------------------------------------------
typedef __attribute__((ext_vector_type(16))) _Float16 v16h;
typedef __attribute__((ext_vector_type(8)))  _Float16 h8;
typedef __attribute__((ext_vector_type(8)))  float    v8f;

__device__ __forceinline__ int lane_id() { return threadIdx.x & 31; }

// A fragment: row-major [M,K] f16, tile (m0,k0), 16x32.
// Lane L (0-15): M=L, K = k0+{0..7} in v0..3, k0+{16..23} in v4..7.
// Lane L (16-31): M=L-16, K = k0+{8..15}, k0+{24..31}.
__device__ __forceinline__ v16h load_afrag(const _Float16* A, int lda, int m0, int k0) {
  int l = lane_id();
  const _Float16* p = A + (size_t)(m0 + (l & 15)) * lda + k0 + ((l >> 4) << 3);
  h8 lo = *(const h8*)(p);
  h8 hi = *(const h8*)(p + 16);
  v16h r;
#pragma unroll
  for (int i = 0; i < 8; ++i) { r[i] = lo[i]; r[i + 8] = hi[i]; }
  return r;
}

// B fragment from TRANSPOSED weights Bt [N,K] f16 (so a column of B is a
// contiguous row of Bt). Lane L (0-15): N=L, K=k0+{0..7},{16..23}; lanes 16-31: K=k0+{8..15},{24..31}.
__device__ __forceinline__ v16h load_bfrag(const _Float16* Bt, int ldb, int n0, int k0) {
  int l = lane_id();
  const _Float16* p = Bt + (size_t)(n0 + (l & 15)) * ldb + k0 + ((l >> 4) << 3);
  h8 lo = *(const h8*)(p);
  h8 hi = *(const h8*)(p + 16);
  v16h r;
#pragma unroll
  for (int i = 0; i < 8; ++i) { r[i] = lo[i]; r[i + 8] = hi[i]; }
  return r;
}

__device__ __forceinline__ v8f wmma_f16(v16h a, v16h b, v8f c) {
  return __builtin_amdgcn_wmma_f32_16x16x32_f16(false, a, false, b, (short)0, c, false, false);
}

// Store one 16x16 f32 accumulator tile to C (ldc = N) with bias/act, col guard.
__device__ __forceinline__ void store_tile(v8f acc, float* C, _Float16* Ch,
                                           const float* bias, int m0, int n0,
                                           int N, int act) {
  int l = lane_id();
  int col = n0 + (l & 15);
  if (col >= N) return;
  float bv = bias ? bias[col] : 0.f;
  int rbase = m0 + ((l >> 4) << 3);
#pragma unroll
  for (int r = 0; r < 8; ++r) {
    float v = acc[r] + bv;
    if (act) v = fmaxf(v, 0.f);
    size_t idx = (size_t)(rbase + r) * N + col;
    if (C)  C[idx]  = v;
    if (Ch) Ch[idx] = (_Float16)v;
  }
}

// ---------------------------------------------------------------------------
// Weight convert+transpose: f32 [K,N] row-major -> f16 [Npad,K], zero padded.
// ---------------------------------------------------------------------------
__global__ void convw_kernel(const float* __restrict__ W, _Float16* __restrict__ Wt,
                             int K, int N, int Npad) {
  int idx = blockIdx.x * blockDim.x + threadIdx.x;
  int tot = Npad * K;
  if (idx >= tot) return;
  int n = idx / K, k = idx - n * K;
  Wt[idx] = (n < N) ? (_Float16)W[(size_t)k * N + n] : (_Float16)0.f;
}

__global__ void f2h_kernel(const float* __restrict__ X, _Float16* __restrict__ Y, int n) {
  int i = blockIdx.x * blockDim.x + threadIdx.x;
  if (i < n) Y[i] = (_Float16)X[i];
}

// ---------------------------------------------------------------------------
// Generic WMMA GEMM, 2x2 register-blocked: each wave computes a 32x32 output
// macro-tile (4 accumulators). Per K-step: 2 A-frags + 2 B-frags -> 4 WMMAs
// (16 FLOP/byte from cache; 4 independent WMMA chains hide XDL latency).
// A [M,K] f16 (M % 32 == 0), Bt [NT2*32, K] f16 zero-padded, C/Ch [M,N].
// ---------------------------------------------------------------------------
__global__ __launch_bounds__(256)
void gemm_f16_kernel(const _Float16* __restrict__ A, const _Float16* __restrict__ Bt,
                     const float* __restrict__ bias,
                     float* __restrict__ C, _Float16* __restrict__ Ch,
                     int M, int N, int K, int NT2, int act) {
  int wid = blockIdx.x * (blockDim.x >> 5) + (threadIdx.x >> 5);
  int MT2 = M >> 5;
  if (wid >= MT2 * NT2) return;          // wave-uniform, EXEC stays full for WMMA
  int mt = wid / NT2, nt = wid - mt * NT2;
  int m0 = mt << 5, n0 = nt << 5;
  v8f a00 = {}, a01 = {}, a10 = {}, a11 = {};
  for (int k0 = 0; k0 < K; k0 += 32) {
    v16h af0 = load_afrag(A, K, m0,      k0);
    v16h af1 = load_afrag(A, K, m0 + 16, k0);
    v16h bf0 = load_bfrag(Bt, K, n0,      k0);
    v16h bf1 = load_bfrag(Bt, K, n0 + 16, k0);
    a00 = wmma_f16(af0, bf0, a00);
    a01 = wmma_f16(af0, bf1, a01);
    a10 = wmma_f16(af1, bf0, a10);
    a11 = wmma_f16(af1, bf1, a11);
  }
  store_tile(a00, C, Ch, bias, m0,      n0,      N, act);
  store_tile(a01, C, Ch, bias, m0,      n0 + 16, N, act);
  store_tile(a10, C, Ch, bias, m0 + 16, n0,      N, act);
  store_tile(a11, C, Ch, bias, m0 + 16, n0 + 16, N, act);
}

// ---------------------------------------------------------------------------
// Fused AVGA kernel: one workgroup per (b,t).
//   vfh  = f16(video tile 49x512)            (LDS, padded to 64 rows)
//   vhh  = f16(relu(vfh @ av_vw + av_vb))     (LDS, WMMA 2x2-blocked, K=512)
//   cont = vhh @ av_v2w + ag[bt,row]          (LDS f32, WMMA, 64x64)
//   z[p] = sum_q tanh(cont[p,q]) * hw[q];  alpha = softmax_p(z)
//   vth[bt,d] = sum_p alpha[p] * vfh[p,d]
// ---------------------------------------------------------------------------
__global__ __launch_bounds__(256)
void avga_kernel(const float* __restrict__ video, const _Float16* __restrict__ av_vwt,
                 const float* __restrict__ av_vb, const _Float16* __restrict__ av_v2wt,
                 const float* __restrict__ ag, const float* __restrict__ hw,
                 _Float16* __restrict__ vth) {
  extern __shared__ char smem[];
  _Float16* vfh = (_Float16*)smem;            // 64*512 f16 = 64 KB
  _Float16* vhh = vfh + 64 * 512;             // 64*512 f16 = 64 KB
  float* cont = (float*)(vhh + 64 * 512);     // 64*64 f32 = 16 KB
  float* zb   = cont + 64 * 64;               // 64
  float* al   = zb + 64;                      // 64
  int bt = blockIdx.x;
  int tid = threadIdx.x;
  const float* vsrc = video + (size_t)bt * 49 * 512;
  for (int i = tid; i < 49 * 512; i += 256) vfh[i] = (_Float16)vsrc[i];
  for (int i = tid; i < 15 * 512; i += 256) vfh[49 * 512 + i] = (_Float16)0.f;
  __syncthreads();

  int wid = tid >> 5, l = lane_id();
  // GEMM1: [64x512] = relu([64x512] @ [512x512] + b)
  // 2x16 macro-tiles (32x32 each) over 8 waves -> 4 per wave, K=512.
  for (int i = 0; i < 4; ++i) {
    int tile = wid * 4 + i;
    int m0 = (tile >> 4) << 5;          // 0 or 32
    int n0 = (tile & 15) << 5;          // 0..480
    v8f a00 = {}, a01 = {}, a10 = {}, a11 = {};
    for (int k0 = 0; k0 < 512; k0 += 32) {
      v16h af0 = load_afrag(vfh, 512, m0,      k0);
      v16h af1 = load_afrag(vfh, 512, m0 + 16, k0);
      v16h bf0 = load_bfrag(av_vwt, 512, n0,      k0);
      v16h bf1 = load_bfrag(av_vwt, 512, n0 + 16, k0);
      a00 = wmma_f16(af0, bf0, a00);
      a01 = wmma_f16(af0, bf1, a01);
      a10 = wmma_f16(af1, bf0, a10);
      a11 = wmma_f16(af1, bf1, a11);
    }
    int rb = (l >> 4) << 3;
#pragma unroll
    for (int r = 0; r < 8; ++r) {
      int c0 = n0 + (l & 15);
      vhh[(m0      + rb + r) * 512 + c0     ] = (_Float16)fmaxf(a00[r] + av_vb[c0],      0.f);
      vhh[(m0      + rb + r) * 512 + c0 + 16] = (_Float16)fmaxf(a01[r] + av_vb[c0 + 16], 0.f);
      vhh[(m0 + 16 + rb + r) * 512 + c0     ] = (_Float16)fmaxf(a10[r] + av_vb[c0],      0.f);
      vhh[(m0 + 16 + rb + r) * 512 + c0 + 16] = (_Float16)fmaxf(a11[r] + av_vb[c0 + 16], 0.f);
    }
  }
  __syncthreads();
  // GEMM2: cont[64x64] = vhh @ av_v2w (N=49 padded), + audio gate broadcast
  for (int i = 0; i < 2; ++i) {
    int tile = wid * 2 + i;
    int m0 = (tile >> 2) << 4, n0 = (tile & 3) << 4;
    v8f acc = {};
    for (int k0 = 0; k0 < 512; k0 += 32)
      acc = wmma_f16(load_afrag(vhh, 512, m0, k0), load_bfrag(av_v2wt, 512, n0, k0), acc);
    int col = n0 + (l & 15);
    int rbase = m0 + ((l >> 4) << 3);
#pragma unroll
    for (int r = 0; r < 8; ++r) {
      int row = rbase + r;
      float a = (row < 49) ? ag[(size_t)bt * 49 + row] : 0.f;
      cont[row * 64 + col] = acc[r] + a;
    }
  }
  __syncthreads();
  if (tid < 49) {
    float s = 0.f;
    for (int q = 0; q < 49; ++q) s += tanhf(cont[tid * 64 + q]) * hw[q];
    zb[tid] = s;
  }
  __syncthreads();
  if (tid == 0) {
    float m = zb[0];
    for (int p = 1; p < 49; ++p) m = fmaxf(m, zb[p]);
    float s = 0.f;
    for (int p = 0; p < 49; ++p) { float e = expf(zb[p] - m); al[p] = e; s += e; }
    float inv = 1.f / s;
    for (int p = 0; p < 49; ++p) al[p] *= inv;
  }
  __syncthreads();
  for (int d = tid; d < 512; d += 256) {
    float s = 0.f;
    for (int p = 0; p < 49; ++p) s += al[p] * (float)vfh[p * 512 + d];
    vth[(size_t)bt * 512 + d] = (_Float16)s;
  }
}

// ---------------------------------------------------------------------------
// LSTM recurrence (one direction). xw already holds x@W_ih + b for all t.
// 16 batch rows per workgroup (batch rows are independent); h kept in LDS f16.
// Per step: g[16,512] = h[16,128] @ whh (WMMA, 1x2-blocked) + xw; gates f32.
// ---------------------------------------------------------------------------
__global__ __launch_bounds__(128)
void lstm_kernel(const float* __restrict__ xw, const _Float16* __restrict__ whht,
                 float* __restrict__ out, _Float16* __restrict__ outh,
                 int dir, int coff) {
  __shared__ _Float16 hsh[16 * 128];
  __shared__ float    csh[16 * 128];
  __shared__ float    gsh[16 * 512];
  int tid = threadIdx.x;
  int b0 = blockIdx.x * 16;
  for (int i = tid; i < 16 * 128; i += 128) { hsh[i] = (_Float16)0.f; csh[i] = 0.f; }
  __syncthreads();
  int wid = tid >> 5, l = lane_id();
  for (int s = 0; s < 10; ++s) {
    int t = dir ? 9 - s : s;
    for (int i = 0; i < 4; ++i) {           // 16 N-macro-tiles over 4 waves
      int n0 = (wid * 4 + i) << 5;
      v8f acc0 = {}, acc1 = {};
      for (int k0 = 0; k0 < 128; k0 += 32) {
        v16h af = load_afrag(hsh, 128, 0, k0);
        acc0 = wmma_f16(af, load_bfrag(whht, 128, n0,      k0), acc0);
        acc1 = wmma_f16(af, load_bfrag(whht, 128, n0 + 16, k0), acc1);
      }
      int col = n0 + (l & 15);
      int rbase = (l >> 4) << 3;
#pragma unroll
      for (int r = 0; r < 8; ++r) {
        int row = rbase + r;
        size_t xb = ((size_t)(b0 + row) * 10 + t) * 512;
        gsh[row * 512 + col     ] = acc0[r] + xw[xb + col];
        gsh[row * 512 + col + 16] = acc1[r] + xw[xb + col + 16];
      }
    }
    __syncthreads();
    for (int i = tid; i < 16 * 128; i += 128) {
      int row = i >> 7, hx = i & 127;
      float gi = gsh[row * 512 + hx];
      float gf = gsh[row * 512 + 128 + hx];
      float gg = gsh[row * 512 + 256 + hx];
      float go = gsh[row * 512 + 384 + hx];
      float si = 1.f / (1.f + expf(-gi));
      float sf = 1.f / (1.f + expf(-gf));
      float so = 1.f / (1.f + expf(-go));
      float c = sf * csh[i] + si * tanhf(gg);
      float h = so * tanhf(c);
      csh[i] = c;
      hsh[i] = (_Float16)h;
      size_t o = ((size_t)(b0 + row) * 10 + t) * 256 + coff + hx;
      out[o] = h;
      outh[o] = (_Float16)h;
    }
    __syncthreads();
  }
}

// ---------------------------------------------------------------------------
// PSP attention per batch element: 10x10 attention, sum-normalize/threshold/
// renormalize both ways, then v_psp = lstm_v + att_v2a@a_b2, a_psp likewise.
// ---------------------------------------------------------------------------
__global__ __launch_bounds__(128)
void psp_kernel(const float* __restrict__ vb2, const float* __restrict__ ab1,
                const float* __restrict__ ab2, const float* __restrict__ vb1,
                const float* __restrict__ lstm_v, const float* __restrict__ lstm_a,
                const float* __restrict__ thrp,
                _Float16* __restrict__ vpsph, _Float16* __restrict__ apsph) {
  __shared__ float att[100], v2a[100], a2v[100];
  int b = blockIdx.x, tid = threadIdx.x;
  float thr = thrp[0];
  const float* V2 = vb2 + (size_t)b * 2560;
  const float* A1 = ab1 + (size_t)b * 2560;
  if (tid < 100) {
    int v = tid / 10, a = tid - v * 10;
    float s = 0.f;
    for (int d = 0; d < 256; ++d) s += V2[v * 256 + d] * A1[a * 256 + d];
    att[tid] = fmaxf(s * 0.0625f, 0.f);   // scale = 1/sqrt(256)
  }
  __syncthreads();
  if (tid < 10) {                          // v->a rows
    float s = 0.f;
    for (int a = 0; a < 10; ++a) s += att[tid * 10 + a];
    float inv = 1.f / (s + 1e-8f);
    float tmp[10]; float s2 = 0.f;
    for (int a = 0; a < 10; ++a) { float x = att[tid * 10 + a] * inv; x = (x > thr) ? x : 0.f; tmp[a] = x; s2 += x; }
    float inv2 = 1.f / (s2 + 1e-8f);
    for (int a = 0; a < 10; ++a) v2a[tid * 10 + a] = tmp[a] * inv2;
  } else if (tid < 20) {                   // a->v rows (columns of att)
    int a = tid - 10;
    float s = 0.f;
    for (int v = 0; v < 10; ++v) s += att[v * 10 + a];
    float inv = 1.f / (s + 1e-8f);
    float tmp[10]; float s2 = 0.f;
    for (int v = 0; v < 10; ++v) { float x = att[v * 10 + a] * inv; x = (x > thr) ? x : 0.f; tmp[v] = x; s2 += x; }
    float inv2 = 1.f / (s2 + 1e-8f);
    for (int v = 0; v < 10; ++v) a2v[a * 10 + v] = tmp[v] * inv2;
  }
  __syncthreads();
  const float* A2 = ab2 + (size_t)b * 2560;
  const float* V1 = vb1 + (size_t)b * 2560;
  const float* LV = lstm_v + (size_t)b * 2560;
  const float* LA = lstm_a + (size_t)b * 2560;
  for (int i = tid; i < 2560; i += 128) {
    int v = i >> 8, d = i & 255;
    float sv = LV[i];
    for (int a = 0; a < 10; ++a) sv += v2a[v * 10 + a] * A2[a * 256 + d];
    vpsph[(size_t)b * 2560 + i] = (_Float16)sv;
    float sa = LA[i];
    for (int u = 0; u < 10; ++u) sa += a2v[v * 10 + u] * V1[u * 256 + d];
    apsph[(size_t)b * 2560 + i] = (_Float16)sa;
  }
}

// ---------------------------------------------------------------------------
// LayerNorm both branches + fuse + cosine similarity. One wave per row (256
// features -> 8 per lane), wave32 shuffle reductions.
// ---------------------------------------------------------------------------
__global__ __launch_bounds__(256)
void lnfuse_kernel(const float* __restrict__ vp2, const float* __restrict__ ap2,
                   const float* __restrict__ g, const float* __restrict__ bb,
                   float* __restrict__ fuse, _Float16* __restrict__ fuseh,
                   float* __restrict__ cross) {
  int row = blockIdx.x * 8 + (threadIdx.x >> 5);
  int l = lane_id();
  const float* vr = vp2 + (size_t)row * 256;
  const float* ar = ap2 + (size_t)row * 256;
  float v[8], a[8];
  float sv = 0.f, sa = 0.f;
#pragma unroll
  for (int i = 0; i < 8; ++i) {
    v[i] = vr[l + 32 * i]; a[i] = ar[l + 32 * i];
    sv += v[i]; sa += a[i];
  }
  for (int o = 16; o > 0; o >>= 1) { sv += __shfl_xor(sv, o, 32); sa += __shfl_xor(sa, o, 32); }
  float mv = sv * (1.f / 256.f), ma = sa * (1.f / 256.f);
  float qv = 0.f, qa = 0.f;
#pragma unroll
  for (int i = 0; i < 8; ++i) { float dv = v[i] - mv, da = a[i] - ma; qv += dv * dv; qa += da * da; }
  for (int o = 16; o > 0; o >>= 1) { qv += __shfl_xor(qv, o, 32); qa += __shfl_xor(qa, o, 32); }
  float rv = rsqrtf(qv * (1.f / 256.f) + 1e-6f);
  float ra = rsqrtf(qa * (1.f / 256.f) + 1e-6f);
  float nv2 = 0.f, na2 = 0.f, dot = 0.f;
#pragma unroll
  for (int i = 0; i < 8; ++i) {
    int c = l + 32 * i;
    float gv = g[c], bv = bb[c];
    float x = (v[i] - mv) * rv * gv + bv;
    float y = (a[i] - ma) * ra * gv + bv;
    float f = 0.5f * (x + y);
    fuse[(size_t)row * 256 + c] = f;
    fuseh[(size_t)row * 256 + c] = (_Float16)f;
    nv2 += x * x; na2 += y * y; dot += x * y;
  }
  for (int o = 16; o > 0; o >>= 1) {
    nv2 += __shfl_xor(nv2, o, 32); na2 += __shfl_xor(na2, o, 32); dot += __shfl_xor(dot, o, 32);
  }
  if (l == 0)
    cross[row] = dot / (fmaxf(sqrtf(nv2), 1e-12f) * fmaxf(sqrtf(na2), 1e-12f));
}

// ---------------------------------------------------------------------------
// Host orchestration
// ---------------------------------------------------------------------------
extern "C" void kernel_launch(void* const* d_in, const int* in_sizes, int n_in,
                              void* d_out, int out_size, void* d_ws, size_t ws_size,
                              hipStream_t stream) {
  (void)in_sizes; (void)n_in; (void)out_size; (void)ws_size;
  const float* audio  = (const float*)d_in[0];
  const float* video  = (const float*)d_in[1];
  const float* thrp   = (const float*)d_in[2];
  const float* fa_w1  = (const float*)d_in[3];
  const float* fa_w2  = (const float*)d_in[4];
  const float* av_aw  = (const float*)d_in[5];
  const float* av_ab  = (const float*)d_in[6];
  const float* av_vw  = (const float*)d_in[7];
  const float* av_vb  = (const float*)d_in[8];
  const float* av_v2w = (const float*)d_in[9];
  const float* av_gw  = (const float*)d_in[10];
  const float* av_hw  = (const float*)d_in[11];
  const float* fv_w1  = (const float*)d_in[12];
  const float* fv_w2  = (const float*)d_in[13];
  const float* la_wih = (const float*)d_in[14];
  const float* la_whh = (const float*)d_in[15];
  const float* la_b   = (const float*)d_in[16];
  const float* lv_wih = (const float*)d_in[17];
  const float* lv_whh = (const float*)d_in[18];
  const float* lv_b   = (const float*)d_in[19];
  const float* vL1    = (const float*)d_in[20];
  const float* vL2    = (const float*)d_in[21];
  const float* aL1    = (const float*)d_in[22];
  const float* aL2    = (const float*)d_in[23];
  const float* vfc    = (const float*)d_in[24];
  const float* afc    = (const float*)d_in[25];
  const float* ln_g   = (const float*)d_in[26];
  const float* ln_b   = (const float*)d_in[27];
  const float* L1w    = (const float*)d_in[28];
  const float* L2w    = (const float*)d_in[29];

  const int BT = 2560;
  char* ws = (char*)d_ws;
  size_t off = 0;
  auto alloc  = [&](size_t bytes) -> char* { size_t o = (off + 255) & ~(size_t)255; off = o + bytes; return ws + o; };
  auto allocH = [&](size_t n) -> _Float16* { return (_Float16*)alloc(n * sizeof(_Float16)); };
  auto allocF = [&](size_t n) -> float*    { return (float*)alloc(n * sizeof(float)); };

  // activations
  _Float16* audioh = allocH((size_t)BT * 128);
  _Float16* tmp1h  = allocH((size_t)BT * 256);
  _Float16* fah    = allocH((size_t)BT * 128);
  _Float16* ahh    = allocH((size_t)BT * 512);
  float*    ag     = allocF((size_t)BT * 49);
  _Float16* vth    = allocH((size_t)BT * 512);
  _Float16* vt1h   = allocH((size_t)BT * 256);
  _Float16* vtfh   = allocH((size_t)BT * 128);
  float* xwaf = allocF((size_t)BT * 512);
  float* xwab = allocF((size_t)BT * 512);
  float* xwvf = allocF((size_t)BT * 512);
  float* xwvb = allocF((size_t)BT * 512);
  float* lstm_a = allocF((size_t)BT * 256);
  float* lstm_v = allocF((size_t)BT * 256);
  _Float16* lstm_ah = allocH((size_t)BT * 256);
  _Float16* lstm_vh = allocH((size_t)BT * 256);
  float* vb1 = allocF((size_t)BT * 256);
  float* vb2 = allocF((size_t)BT * 256);
  float* ab1 = allocF((size_t)BT * 256);
  float* ab2 = allocF((size_t)BT * 256);
  _Float16* vpsph = allocH((size_t)BT * 256);
  _Float16* apsph = allocH((size_t)BT * 256);
  float* vp2 = allocF((size_t)BT * 256);
  float* ap2 = allocF((size_t)BT * 256);
  _Float16* fuseh = allocH((size_t)BT * 256);
  _Float16* g1h   = allocH((size_t)BT * 64);
  // f16 transposed weights [Npad,K]
  _Float16* w_fa1t = allocH(256 * 128);
  _Float16* w_fa2t = allocH(128 * 256);
  _Float16* w_avat = allocH(512 * 128);
  _Float16* w_avvt = allocH(512 * 512);
  _Float16* w_v2t  = allocH(64 * 512);
  _Float16* w_gwt  = allocH(64 * 512);
  _Float16* w_fv1t = allocH(256 * 512);
  _Float16* w_fv2t = allocH(128 * 256);
  _Float16* w_awih0 = allocH(512 * 128);
  _Float16* w_awih1 = allocH(512 * 128);
  _Float16* w_awhh0 = allocH(512 * 128);
  _Float16* w_awhh1 = allocH(512 * 128);
  _Float16* w_vwih0 = allocH(512 * 128);
  _Float16* w_vwih1 = allocH(512 * 128);
  _Float16* w_vwhh0 = allocH(512 * 128);
  _Float16* w_vwhh1 = allocH(512 * 128);
  _Float16* w_vL1t = allocH(256 * 256);
  _Float16* w_vL2t = allocH(256 * 256);
  _Float16* w_aL1t = allocH(256 * 256);
  _Float16* w_aL2t = allocH(256 * 256);
  _Float16* w_vfct = allocH(256 * 256);
  _Float16* w_afct = allocH(256 * 256);
  _Float16* w_L1t  = allocH(64 * 256);
  _Float16* w_L2t  = allocH(32 * 64);

  float* out_fuse  = (float*)d_out;
  float* out_pred  = out_fuse + (size_t)BT * 256;
  float* out_cross = out_pred + (size_t)BT * 29;

  auto convw = [&](const float* W, _Float16* Wt, int K, int N, int Np) {
    int tot = Np * K;
    convw_kernel<<<(tot + 255) / 256, 256, 0, stream>>>(W, Wt, K, N, Np);
  };
  auto gemm = [&](const _Float16* A, const _Float16* Bt, const float* bias,
                  float* C, _Float16* Ch, int M, int N, int K, int Np, int act) {
    int NT2 = Np >> 5;
    int tiles = (M >> 5) * NT2;
    gemm_f16_kernel<<<(tiles + 7) / 8, 256, 0, stream>>>(A, Bt, bias, C, Ch, M, N, K, NT2, act);
  };

  // ---- weight conversion (f32 [K,N] -> f16 [Npad,K]) ----
  convw(fa_w1, w_fa1t, 128, 256, 256);
  convw(fa_w2, w_fa2t, 256, 128, 128);
  convw(av_aw, w_avat, 128, 512, 512);
  convw(av_vw, w_avvt, 512, 512, 512);
  convw(av_v2w, w_v2t, 512, 49, 64);
  convw(av_gw,  w_gwt, 512, 49, 64);
  convw(fv_w1, w_fv1t, 512, 256, 256);
  convw(fv_w2, w_fv2t, 256, 128, 128);
  convw(la_wih,                w_awih0, 128, 512, 512);
  convw(la_wih + 128 * 512,    w_awih1, 128, 512, 512);
  convw(la_whh,                w_awhh0, 128, 512, 512);
  convw(la_whh + 128 * 512,    w_awhh1, 128, 512, 512);
  convw(lv_wih,                w_vwih0, 128, 512, 512);
  convw(lv_wih + 128 * 512,    w_vwih1, 128, 512, 512);
  convw(lv_whh,                w_vwhh0, 128, 512, 512);
  convw(lv_whh + 128 * 512,    w_vwhh1, 128, 512, 512);
  convw(vL1, w_vL1t, 256, 256, 256);
  convw(vL2, w_vL2t, 256, 256, 256);
  convw(aL1, w_aL1t, 256, 256, 256);
  convw(aL2, w_aL2t, 256, 256, 256);
  convw(vfc, w_vfct, 256, 256, 256);
  convw(afc, w_afct, 256, 256, 256);
  convw(L1w, w_L1t, 256, 64, 64);
  convw(L2w, w_L2t, 64, 29, 32);

  // ---- audio path: fa = (audio @ fa_w1) @ fa_w2 ----
  f2h_kernel<<<(BT * 128 + 255) / 256, 256, 0, stream>>>(audio, audioh, BT * 128);
  gemm(audioh, w_fa1t, nullptr, nullptr, tmp1h, BT, 256, 128, 256, 0);
  gemm(tmp1h,  w_fa2t, nullptr, nullptr, fah,   BT, 128, 256, 128, 0);

  // ---- audio gate: a_h = relu(fa @ av_aw + b);  ag = a_h @ av_gw ----
  gemm(fah, w_avat, av_ab, nullptr, ahh, BT, 512, 128, 512, 1);
  gemm(ahh, w_gwt, nullptr, ag, nullptr, BT, 49, 512, 64, 0);

  // ---- fused AVGA (big WMMA stage) ----
  size_t avga_smem = (size_t)64 * 512 * 2 * 2 + (size_t)64 * 64 * 4 + 128 * 4;
  avga_kernel<<<BT, 256, avga_smem, stream>>>(video, w_avvt, av_vb, w_v2t, ag, av_hw, vth);

  // ---- video feature projection ----
  gemm(vth,  w_fv1t, nullptr, nullptr, vt1h, BT, 256, 512, 256, 0);
  gemm(vt1h, w_fv2t, nullptr, nullptr, vtfh, BT, 128, 256, 128, 0);

  // ---- LSTM input projections (x@W_ih + b for all t) ----
  gemm(fah,  w_awih0, la_b,       xwaf, nullptr, BT, 512, 128, 512, 0);
  gemm(fah,  w_awih1, la_b + 512, xwab, nullptr, BT, 512, 128, 512, 0);
  gemm(vtfh, w_vwih0, lv_b,       xwvf, nullptr, BT, 512, 128, 512, 0);
  gemm(vtfh, w_vwih1, lv_b + 512, xwvb, nullptr, BT, 512, 128, 512, 0);

  // ---- bidirectional LSTM recurrences ----
  lstm_kernel<<<16, 128, 0, stream>>>(xwaf, w_awhh0, lstm_a, lstm_ah, 0, 0);
  lstm_kernel<<<16, 128, 0, stream>>>(xwab, w_awhh1, lstm_a, lstm_ah, 1, 128);
  lstm_kernel<<<16, 128, 0, stream>>>(xwvf, w_vwhh0, lstm_v, lstm_vh, 0, 0);
  lstm_kernel<<<16, 128, 0, stream>>>(xwvb, w_vwhh1, lstm_v, lstm_vh, 1, 128);

  // ---- PSP branches ----
  gemm(lstm_vh, w_vL1t, nullptr, vb1, nullptr, BT, 256, 256, 256, 1);
  gemm(lstm_vh, w_vL2t, nullptr, vb2, nullptr, BT, 256, 256, 256, 1);
  gemm(lstm_ah, w_aL1t, nullptr, ab1, nullptr, BT, 256, 256, 256, 1);
  gemm(lstm_ah, w_aL2t, nullptr, ab2, nullptr, BT, 256, 256, 256, 1);
  psp_kernel<<<256, 128, 0, stream>>>(vb2, ab1, ab2, vb1, lstm_v, lstm_a, thrp, vpsph, apsph);
  gemm(vpsph, w_vfct, nullptr, vp2, nullptr, BT, 256, 256, 256, 1);
  gemm(apsph, w_afct, nullptr, ap2, nullptr, BT, 256, 256, 256, 1);

  // ---- LayerNorm + fuse + cosine ----
  lnfuse_kernel<<<320, 256, 0, stream>>>(vp2, ap2, ln_g, ln_b, out_fuse, fuseh, out_cross);

  // ---- classifier head ----
  gemm(fuseh, w_L1t, nullptr, nullptr, g1h, BT, 64, 256, 64, 1);
  gemm(g1h,   w_L2t, nullptr, out_pred, nullptr, BT, 29, 64, 32, 0);
}
